// NeighborhoodAttention1D_10368051053041
// MI455X (gfx1250) — matrix operand order repaired and checked by reading, not compiled, not run on hardware
//
#include <hip/hip_runtime.h>

// ---------------- problem constants ----------------
#define LSEQ   4096
#define CH     1024
#define HEADS  16
#define HDIM   64
#define NK     13
#define NS     6
#define NQKV   3072
#define BSZ    2
#define MROWS  (BSZ * LSEQ)       // 8192
#define SCALE  0.125f             // 64^-0.5

typedef __attribute__((ext_vector_type(16))) __bf16 v16bf;
typedef __attribute__((ext_vector_type(8)))  float  v8f;
typedef __attribute__((ext_vector_type(4))) unsigned int ui32x4;
typedef __attribute__((ext_vector_type(8))) int i32x8;
typedef __attribute__((ext_vector_type(4))) int i32x4;

union Frag16 { v16bf v; unsigned int u[8]; };

#if __has_builtin(__builtin_amdgcn_tensor_load_to_lds)
  #define USE_TDM 1
  #if defined(__clang_major__) && (__clang_major__ >= 23)
    // amdgpu-toolchain (clang-23): 6-arg form with extra i32x8 group
    #define TDM_LOAD(g0, g1, g2, g3) \
      __builtin_amdgcn_tensor_load_to_lds((g0), (g1), (g2), (g3), (i32x8){0,0,0,0,0,0,0,0}, 0)
  #else
    // ROCm 7.2 (clang-22): 5-arg form
    #define TDM_LOAD(g0, g1, g2, g3) \
      __builtin_amdgcn_tensor_load_to_lds((g0), (g1), (g2), (g3), 0)
  #endif
#else
  #define USE_TDM 0
#endif

static __device__ __forceinline__ unsigned short f32_to_bf16(float f) {
  unsigned int u = __float_as_uint(f);
  unsigned int r = 0x7FFFu + ((u >> 16) & 1u);
  u += r;
  return (unsigned short)(u >> 16);
}

static __device__ __forceinline__ void lds_fence() {
  asm volatile("s_wait_dscnt 0" ::: "memory");
}

// A-fragment K offsets for 16-bit 16x16x32: VGPR p, lane half hf
static __device__ __forceinline__ int kaOff(int p, int hf) {
  return 2 * p + 16 * hf;
}
// B-fragment K offsets: VGPR p, lane half hf
static __device__ __forceinline__ int kbOff(int p, int hf) {
  return 2 * p + 16 * hf;
}

// ---------------- f32 -> bf16 conversion ----------------
__global__ __launch_bounds__(256) void cvt_bf16_kernel(const float* __restrict__ in,
                                                       unsigned short* __restrict__ out,
                                                       int n) {
  int i = blockIdx.x * 256 + threadIdx.x;
  if (i < n) out[i] = f32_to_bf16(in[i]);
}

// ---------------- QKV GEMM: (8192x1024) x (1024x3072) ----------------
__global__ __launch_bounds__(256) void qkv_gemm_kernel(const unsigned short* __restrict__ xb,
                                                       const unsigned short* __restrict__ wb,
                                                       const float* __restrict__ bias,
                                                       unsigned short* __restrict__ qb,
                                                       unsigned short* __restrict__ kbuf,
                                                       unsigned short* __restrict__ vbuf) {
  const int lane = threadIdx.x & 31;
  const int wave = threadIdx.x >> 5;          // 0..7
  const int wm = wave >> 1;                   // 0..3
  const int wn = wave & 1;                    // 0..1
  const int m0 = (blockIdx.y * 4 + wm) * 16;
  const int n0 = (blockIdx.x * 2 + wn) * 16;
  const int hf = lane >> 4;
  const int mr = lane & 15;

  const unsigned short* arow = xb + (size_t)(m0 + mr) * CH;
  const unsigned short* bcol = wb + (size_t)(n0 + mr) * CH;

  v8f acc = {};
  for (int kt = 0; kt < CH; kt += 32) {
    Frag16 A, B;
#pragma unroll
    for (int p = 0; p < 8; ++p) {
      A.u[p] = *(const unsigned int*)(arow + kt + kaOff(p, hf));
      B.u[p] = *(const unsigned int*)(bcol + kt + kbOff(p, hf));
    }
    __builtin_prefetch(arow + kt + 64, 0, 0);
    __builtin_prefetch(bcol + kt + 64, 0, 0);
    acc = __builtin_amdgcn_wmma_f32_16x16x32_bf16(false, A.v, false, B.v,
                                                  (short)0, acc, false, false);
  }

  const int n = n0 + (lane & 15);
  const int which = n >> 10;                  // 0=q 1=k 2=v
  const int nc = n & 1023;
  const int h = nc >> 6, d = nc & 63;
  const float bv = bias[n];
  unsigned short* dst = (which == 0) ? qb : (which == 1) ? kbuf : vbuf;
#pragma unroll
  for (int r = 0; r < 8; ++r) {
    int m = m0 + r + 8 * hf;
    int b = m >> 12;                          // /4096
    int l = m & (LSEQ - 1);
    float val = acc[r] + bv;
    if (which == 0) val *= SCALE;
    dst[(size_t)(((b * HEADS + h) * LSEQ) + l) * HDIM + d] = f32_to_bf16(val);
  }
}

// ---------------- windowed neighborhood attention ----------------
struct __align__(16) WaveShm {
  float S[16][32];
  unsigned short P[16][32];
  unsigned short Vw[32][64];  // row-major V window: Vw[kk][n] = v[w0+kk][n]
};

__global__ __launch_bounds__(128) void natten_kernel(const unsigned short* __restrict__ qb,
                                                     const unsigned short* __restrict__ kbuf,
                                                     const unsigned short* __restrict__ vbuf,
                                                     const float* __restrict__ rpb,
                                                     unsigned short* __restrict__ ab) {
  __shared__ WaveShm shm[4];
  const int lane = threadIdx.x & 31;
  const int wave = threadIdx.x >> 5;
  WaveShm& S = shm[wave];

  const int tile = blockIdx.x * 4 + wave;     // 0..8191
  const int lt = tile & 255;                  // L tile
  const int h = (tile >> 8) & (HEADS - 1);
  const int b = tile >> 12;
  const int bh = b * HEADS + h;
  const int l0 = lt * 16;
  int w0 = l0 - NS;
  if (w0 < 0) w0 = 0;
  if (w0 > LSEQ - NK) w0 = LSEQ - NK;

  const int hf = lane >> 4;
  const int mr = lane & 15;

  // ---- stage V window [32 rows][64 cols] bf16 into LDS ----
#if USE_TDM
  {
    // Tensor DMA descriptor (D#), ISA ch.8: 2D tensor, data_size=2B,
    // tile 64(x) x 32(y), row stride 64; rows past (LSEQ - w0) read as zero.
    unsigned long long ga =
        (unsigned long long)(const void*)(vbuf + (size_t)(bh * LSEQ + w0) * HDIM);
    unsigned ldsAddr = (unsigned)(size_t)(void*)&S.Vw[0][0];  // flat low 32 = LDS offset
    unsigned td1 = (unsigned)(LSEQ - w0);                     // rows available
    ui32x4 g0;
    g0[0] = 1u;                                        // count=1, user descriptor
    g0[1] = ldsAddr;                                   // lds_addr
    g0[2] = (unsigned)(ga & 0xFFFFFFFFu);              // global_addr[31:0]
    g0[3] = (unsigned)((ga >> 32) & 0x01FFFFFFu)       // global_addr[56:32]
          | (2u << 30);                                // type=2 (image)
    i32x8 g1;
    g1[0] = (int)(1u << 16);                           // data_size=1 (2 bytes)
    g1[1] = (int)(64u << 16);                          // tensor_dim0=64 (low16 @[31:16])
    g1[2] = (int)((td1 & 0xFFFFu) << 16);              // dim0 hi16=0, tensor_dim1 low16
    g1[3] = (int)((td1 >> 16) & 0xFFFFu)               // tensor_dim1 hi16
          | (int)(64u << 16);                          // tile_dim0=64
    g1[4] = 32;                                        // tile_dim1=32, tile_dim2=0
    g1[5] = 64;                                        // tensor_dim0_stride low32
    g1[6] = (int)(64u << 16);                          // stride0 hi16=0, stride1 low16=64
    g1[7] = 0;                                         // stride1 hi
    i32x4 gz = {0, 0, 0, 0};
    TDM_LOAD(g0, g1, gz, gz);
  }
#else
  for (int i = lane; i < 32 * 64; i += 32) {
    int n = i & 63;
    int kk = i >> 6;
    int row = w0 + kk;
    S.Vw[kk][n] = (row < LSEQ)
        ? vbuf[(size_t)(bh * LSEQ + row) * HDIM + n]
        : (unsigned short)0;
  }
#endif

  // ---- S = Q (16x64) * K^T (64x28pad32) : 4 WMMA (overlaps with TDM) ----
  const unsigned short* qrow = qb + (size_t)(bh * LSEQ + l0 + mr) * HDIM;
  int kr0 = w0 + (lane & 15);        if (kr0 > LSEQ - 1) kr0 = LSEQ - 1;
  int kr1 = w0 + 16 + (lane & 15);   if (kr1 > LSEQ - 1) kr1 = LSEQ - 1;
  const unsigned short* krow0 = kbuf + (size_t)(bh * LSEQ + kr0) * HDIM;
  const unsigned short* krow1 = kbuf + (size_t)(bh * LSEQ + kr1) * HDIM;

  v8f s0 = {}, s1 = {};
  for (int kt = 0; kt < HDIM; kt += 32) {
    Frag16 A, B0, B1;
#pragma unroll
    for (int p = 0; p < 8; ++p) {
      A.u[p]  = *(const unsigned int*)(qrow  + kt + kaOff(p, hf));
      B0.u[p] = *(const unsigned int*)(krow0 + kt + kbOff(p, hf));
      B1.u[p] = *(const unsigned int*)(krow1 + kt + kbOff(p, hf));
    }
    s0 = __builtin_amdgcn_wmma_f32_16x16x32_bf16(false, A.v, false, B0.v, (short)0, s0, false, false);
    s1 = __builtin_amdgcn_wmma_f32_16x16x32_bf16(false, A.v, false, B1.v, (short)0, s1, false, false);
  }
#pragma unroll
  for (int r = 0; r < 8; ++r) {
    S.S[r + 8 * hf][lane & 15]        = s0[r];
    S.S[r + 8 * hf][16 + (lane & 15)] = s1[r];
  }
  lds_fence();

  // ---- per-row mask + rpb + softmax (lanes 0..15, one row each) ----
  if (lane < 16) {
    const int m = lane;
    const int l = l0 + m;
    int ni = l - NS; if (ni < 0) ni = 0; if (ni > LSEQ - NK) ni = LSEQ - NK;
    const int off = ni - w0;                          // window column of j=0
    int pi = NS + ((l < NS) ? (NS - l) : 0) + ((l + NS >= LSEQ) ? (LSEQ - l - 1 - NS) : 0);
    const float* rpbh = rpb + h * (2 * NK - 1);
    float e[NK];
    float mx = -3.0e38f;
#pragma unroll
    for (int j = 0; j < NK; ++j) {
      float sc = S.S[m][off + j] + rpbh[pi + j];
      e[j] = sc;
      if (sc > mx) mx = sc;
    }
    float sum = 0.f;
#pragma unroll
    for (int j = 0; j < NK; ++j) { e[j] = __expf(e[j] - mx); sum += e[j]; }
    float inv = 1.f / sum;
#pragma unroll
    for (int c = 0; c < 32; ++c) S.P[m][c] = 0;
#pragma unroll
    for (int j = 0; j < NK; ++j) S.P[m][off + j] = f32_to_bf16(e[j] * inv);
  }
  lds_fence();

#if USE_TDM
  __builtin_amdgcn_s_wait_tensorcnt(0);   // V window resident in LDS
#endif

  // ---- O = P (16x32) * Vw-as-B (32x64) : 4 WMMA ----
  Frag16 Pa;
#pragma unroll
  for (int p = 0; p < 8; ++p)
    Pa.u[p] = *(const unsigned int*)(&S.P[mr][kaOff(p, hf)]);

#pragma unroll
  for (int nt = 0; nt < 4; ++nt) {
    Frag16 Bv;
    const int n = nt * 16 + (lane & 15);
#pragma unroll
    for (int p = 0; p < 8; ++p) {
      const int kk = 2 * p + 16 * hf;
      unsigned lo = S.Vw[kk][n];
      unsigned hi = S.Vw[kk + 1][n];
      Bv.u[p] = lo | (hi << 16);
    }
    v8f o = {};
    o = __builtin_amdgcn_wmma_f32_16x16x32_bf16(false, Pa.v, false, Bv.v, (short)0, o, false, false);
#pragma unroll
    for (int r = 0; r < 8; ++r) {
      int l = l0 + r + 8 * hf;
      ab[(size_t)(b * LSEQ + l) * CH + h * HDIM + n] = f32_to_bf16(o[r]);
    }
  }
}

// ---------------- proj GEMM: (8192x1024) x (1024x1024) + bias -> f32 out ----------------
__global__ __launch_bounds__(256) void proj_gemm_kernel(const unsigned short* __restrict__ ab,
                                                        const unsigned short* __restrict__ wb,
                                                        const float* __restrict__ bias,
                                                        float* __restrict__ out) {
  const int lane = threadIdx.x & 31;
  const int wave = threadIdx.x >> 5;
  const int wm = wave >> 1;
  const int wn = wave & 1;
  const int m0 = (blockIdx.y * 4 + wm) * 16;
  const int n0 = (blockIdx.x * 2 + wn) * 16;
  const int hf = lane >> 4;
  const int mr = lane & 15;

  const unsigned short* arow = ab + (size_t)(m0 + mr) * CH;
  const unsigned short* bcol = wb + (size_t)(n0 + mr) * CH;

  v8f acc = {};
  for (int kt = 0; kt < CH; kt += 32) {
    Frag16 A, B;
#pragma unroll
    for (int p = 0; p < 8; ++p) {
      A.u[p] = *(const unsigned int*)(arow + kt + kaOff(p, hf));
      B.u[p] = *(const unsigned int*)(bcol + kt + kbOff(p, hf));
    }
    __builtin_prefetch(arow + kt + 64, 0, 0);
    __builtin_prefetch(bcol + kt + 64, 0, 0);
    acc = __builtin_amdgcn_wmma_f32_16x16x32_bf16(false, A.v, false, B.v,
                                                  (short)0, acc, false, false);
  }

  const int n = n0 + (lane & 15);
  const float bv = bias[n];
#pragma unroll
  for (int r = 0; r < 8; ++r) {
    int m = m0 + r + 8 * hf;
    out[(size_t)m * CH + n] = acc[r] + bv;
  }
}

// ---------------- host launch ----------------
extern "C" void kernel_launch(void* const* d_in, const int* in_sizes, int n_in,
                              void* d_out, int out_size, void* d_ws, size_t ws_size,
                              hipStream_t stream) {
  const float* x      = (const float*)d_in[0];   // [2,4096,1024]
  const float* qkv_w  = (const float*)d_in[1];   // [3072,1024]
  const float* qkv_b  = (const float*)d_in[2];   // [3072]
  const float* rpb    = (const float*)d_in[3];   // [16,25]
  const float* proj_w = (const float*)d_in[4];   // [1024,1024]
  const float* proj_b = (const float*)d_in[5];   // [1024]
  float* out = (float*)d_out;

  size_t off = 0;
  auto alloc = [&](size_t bytes) -> void* {
    void* p = (char*)d_ws + off;
    off = (off + bytes + 255) & ~(size_t)255;
    return p;
  };
  unsigned short* xb    = (unsigned short*)alloc((size_t)MROWS * CH * 2);
  unsigned short* wqkvb = (unsigned short*)alloc((size_t)NQKV * CH * 2);
  unsigned short* wprjb = (unsigned short*)alloc((size_t)CH * CH * 2);
  unsigned short* qb    = (unsigned short*)alloc((size_t)BSZ * HEADS * LSEQ * HDIM * 2);
  unsigned short* kb    = (unsigned short*)alloc((size_t)BSZ * HEADS * LSEQ * HDIM * 2);
  unsigned short* vb    = (unsigned short*)alloc((size_t)BSZ * HEADS * LSEQ * HDIM * 2);
  unsigned short* attnb = (unsigned short*)alloc((size_t)MROWS * CH * 2);
  (void)ws_size; (void)in_sizes; (void)n_in; (void)out_size;

  // 1) convert x / qkv_w / proj_w to bf16
  {
    int n1 = MROWS * CH;
    cvt_bf16_kernel<<<(n1 + 255) / 256, 256, 0, stream>>>(x, xb, n1);
    int n2 = NQKV * CH;
    cvt_bf16_kernel<<<(n2 + 255) / 256, 256, 0, stream>>>(qkv_w, wqkvb, n2);
    int n3 = CH * CH;
    cvt_bf16_kernel<<<(n3 + 255) / 256, 256, 0, stream>>>(proj_w, wprjb, n3);
  }

  // 2) QKV projection (WMMA bf16)
  {
    dim3 grid(NQKV / 32, MROWS / 64);
    qkv_gemm_kernel<<<grid, 256, 0, stream>>>(xb, wqkvb, qkv_b, qb, kb, vb);
  }

  // 3) windowed neighborhood attention (WMMA bf16 + TDM async V-tile)
  {
    int tiles = BSZ * HEADS * (LSEQ / 16);   // 8192 waves
    natten_kernel<<<tiles / 4, 128, 0, stream>>>(qb, kb, vb, rpb, attnb);
  }

  // 4) output projection (WMMA bf16, f32 out + bias)
  {
    dim3 grid(CH / 32, MROWS / 64);
    proj_gemm_kernel<<<grid, 256, 0, stream>>>(attnb, wprjb, proj_b, out);
  }
}
